// MultiSimilarityLoss_45509473469192
// MI455X (gfx1250) — matrix operand order repaired and checked by reading, not compiled, not run on hardware
//
#include <hip/hip_runtime.h>
#include <hip/hip_bf16.h>
#include <cstdint>

typedef __attribute__((ext_vector_type(16))) _Float16 v16h;
typedef __attribute__((ext_vector_type(4)))  _Float16 v4h;
typedef __attribute__((ext_vector_type(8)))  float    v8f;

#define BATCH 4096
#define DIM   1024
#define APAD  8                  // pad rows by 8 halves -> bank-conflict-free
#define ASTRIDE (DIM + APAD)     // 1032 halves = 2064 B = 516 dwords (= 4-bank shift/row)
#define F_THRESH 0.5f
#define F_MARGIN 0.1f
#define F_SPOS   2.0f
#define F_SNEG   40.0f
#define F_EPS    1e-5f

// ---------------------------------------------------------------------------
// Zero the scalar output (harness poisons d_out).
// ---------------------------------------------------------------------------
__global__ void ms_zero_out(float* out) {
    if (threadIdx.x == 0) out[0] = 0.0f;
}

// ---------------------------------------------------------------------------
// Row-normalize f32 features -> f16 (one block per row, 256 threads,
// exactly one float4 per thread).
// ---------------------------------------------------------------------------
__global__ __launch_bounds__(256) void ms_normalize_f16(
    const float* __restrict__ feats, _Float16* __restrict__ Fh) {
    __shared__ float red[256];
    const int row = blockIdx.x;
    const int tid = threadIdx.x;

    float4 v = ((const float4*)(feats + (size_t)row * DIM))[tid];
    red[tid] = v.x * v.x + v.y * v.y + v.z * v.z + v.w * v.w;
    __syncthreads();
    for (int s = 128; s > 0; s >>= 1) {
        if (tid < s) red[tid] += red[tid + s];
        __syncthreads();
    }
    const float rn = rsqrtf(red[0]);
    v4h h = { (_Float16)(v.x * rn), (_Float16)(v.y * rn),
              (_Float16)(v.z * rn), (_Float16)(v.w * rn) };
    *(v4h*)(Fh + (size_t)row * DIM + (size_t)tid * 4) = h;
}

// ---------------------------------------------------------------------------
// sim = Fh * Fh^T via v_wmma_f32_16x16x32_f16.
// Grid: (BATCH/64, 2); block: 256 threads = 8 waves.
// Each block: 64-row strip x 2048-col half. A-strip staged in ~129 KB LDS
// with +8-half row padding (row stride = 516 dwords => lane l16 starts at
// bank 4*l16: each 16-lane group covers all 64 banks, conflict-free).
// Each wave: 4 M-subtiles x 2 concurrent N-tiles x 8 j-steps.
// Per k-step: 4 A-fragments (8 ds_b128, small k-immediate only) +
// 2 B-fragments (4 global_b128) feed 8 WMMAs.
// Fragment layouts used here (same contiguous k order for A and B):
//   A 16x32 f16: lane M = lane%16; K = (lane/16)*16 + {0..15}
//   B 32x16 f16: lane N = lane%16; K = (lane/16)*16 + {0..15}   (B col = F row)
//   C 16x16 f32: reg r, lane l -> (M = r + 8*(l/16), N = l%16)
// ---------------------------------------------------------------------------
__global__ __launch_bounds__(256) void ms_sim_gemm(
    const _Float16* __restrict__ Fh, float* __restrict__ sim) {
    __shared__ _Float16 As[64 * ASTRIDE];   // 132096 B (< 320 KB/WGP)

    const int rowBase = blockIdx.x * 64;
    const int nBase   = blockIdx.y * (BATCH / 2);

    // Stage the 64-row A-strip into padded LDS rows, 16 B per copy.
    for (int i = threadIdx.x; i < 64 * (DIM / 8); i += 256) {
        const int r = i >> 7;          // DIM/8 = 128 uint4 per row
        const int c = i & 127;
        ((uint4*)(As + (size_t)r * ASTRIDE))[c] =
            ((const uint4*)(Fh + (size_t)(rowBase + r) * DIM))[c];
    }
    __syncthreads();

    const int lane = threadIdx.x & 31;
    const int wave = threadIdx.x >> 5;
    const int half = lane >> 4;   // 0 or 1
    const int l16  = lane & 15;

    // Loop-invariant per-subtile lane pointers: inner loop addresses are
    // base + k*2 bytes only (fits the DS 16-bit offset immediate).
    const _Float16* a0 = As + (size_t)(l16)      * ASTRIDE + half * 16;
    const _Float16* a1 = As + (size_t)(16 + l16) * ASTRIDE + half * 16;
    const _Float16* a2 = As + (size_t)(32 + l16) * ASTRIDE + half * 16;
    const _Float16* a3 = As + (size_t)(48 + l16) * ASTRIDE + half * 16;

    for (int j = 0; j < 8; ++j) {
        const int n0 = nBase + wave * 16 + j * 256;
        const int n1 = n0 + 128;
        const _Float16* bp0 = Fh + (size_t)(n0 + l16) * DIM + half * 16;
        const _Float16* bp1 = Fh + (size_t)(n1 + l16) * DIM + half * 16;

        v8f acc[8];
#pragma unroll
        for (int s = 0; s < 8; ++s) acc[s] = (v8f){};

#pragma unroll 2
        for (int k = 0; k < DIM; k += 32) {
            const v16h b0 = *(const v16h*)(bp0 + k);
            const v16h b1 = *(const v16h*)(bp1 + k);

            v16h af0, af1, af2, af3;
            ((uint4*)&af0)[0] = *(const uint4*)(a0 + k);
            ((uint4*)&af0)[1] = *(const uint4*)(a0 + k + 8);
            ((uint4*)&af1)[0] = *(const uint4*)(a1 + k);
            ((uint4*)&af1)[1] = *(const uint4*)(a1 + k + 8);
            ((uint4*)&af2)[0] = *(const uint4*)(a2 + k);
            ((uint4*)&af2)[1] = *(const uint4*)(a2 + k + 8);
            ((uint4*)&af3)[0] = *(const uint4*)(a3 + k);
            ((uint4*)&af3)[1] = *(const uint4*)(a3 + k + 8);

            acc[0] = __builtin_amdgcn_wmma_f32_16x16x32_f16(
                         false, af0, false, b0, (short)0, acc[0], false, false);
            acc[1] = __builtin_amdgcn_wmma_f32_16x16x32_f16(
                         false, af1, false, b0, (short)0, acc[1], false, false);
            acc[2] = __builtin_amdgcn_wmma_f32_16x16x32_f16(
                         false, af2, false, b0, (short)0, acc[2], false, false);
            acc[3] = __builtin_amdgcn_wmma_f32_16x16x32_f16(
                         false, af3, false, b0, (short)0, acc[3], false, false);
            acc[4] = __builtin_amdgcn_wmma_f32_16x16x32_f16(
                         false, af0, false, b1, (short)0, acc[4], false, false);
            acc[5] = __builtin_amdgcn_wmma_f32_16x16x32_f16(
                         false, af1, false, b1, (short)0, acc[5], false, false);
            acc[6] = __builtin_amdgcn_wmma_f32_16x16x32_f16(
                         false, af2, false, b1, (short)0, acc[6], false, false);
            acc[7] = __builtin_amdgcn_wmma_f32_16x16x32_f16(
                         false, af3, false, b1, (short)0, acc[7], false, false);
        }

#pragma unroll
        for (int t = 0; t < 2; ++t) {
            const int n = (t == 0) ? n0 : n1;
#pragma unroll
            for (int s = 0; s < 4; ++s) {
                float* c = sim + (size_t)(rowBase + s * 16 + half * 8) * BATCH
                               + n + l16;
#pragma unroll
                for (int r = 0; r < 8; ++r) {
                    c[(size_t)r * BATCH] = acc[t * 4 + s][r];
                }
            }
        }
    }
}

// ---------------------------------------------------------------------------
// Per-row hard mining + loss. One block per row, 256 threads.
// Pass 1: min over positives / max over negatives (LDS tree reduce).
// Pass 2: selected exp-sums + validity flags; thread 0 atomically
// accumulates the row loss into the scalar output.
// ---------------------------------------------------------------------------
__global__ __launch_bounds__(256) void ms_row_loss(
    const float* __restrict__ sim, const long long* __restrict__ labels,
    float* __restrict__ out) {
    __shared__ int   slab[BATCH];    // 16 KB label cache
    __shared__ float redA[256];
    __shared__ float redB[256];
    __shared__ int   redI[256];

    const int tid = threadIdx.x;
    const int row = blockIdx.x;

    for (int i = tid; i < BATCH; i += 256) slab[i] = (int)labels[i];
    __syncthreads();

    const int lab = slab[row];
    const float* s = sim + (size_t)row * BATCH;

    // -------- pass 1: mining thresholds --------
    float minPos =  3.4e38f;
    float maxNeg = -3.4e38f;
    for (int i = tid; i < BATCH; i += 256) {
        const float v    = s[i];
        const bool  same = (slab[i] == lab);
        if (same && (v < 1.0f - F_EPS)) minPos = fminf(minPos, v);
        if (!same)                      maxNeg = fmaxf(maxNeg, v);
    }
    redA[tid] = minPos;
    redB[tid] = maxNeg;
    __syncthreads();
    for (int st = 128; st > 0; st >>= 1) {
        if (tid < st) {
            redA[tid] = fminf(redA[tid], redA[tid + st]);
            redB[tid] = fmaxf(redB[tid], redB[tid + st]);
        }
        __syncthreads();
    }
    minPos = redA[0];
    maxNeg = redB[0];
    __syncthreads();   // before reusing redA/redB

    // -------- pass 2: selected exp sums --------
    float posSum = 0.0f, negSum = 0.0f;
    int any = 0;
    for (int i = tid; i < BATCH; i += 256) {
        const float v    = s[i];
        const bool  same = (slab[i] == lab);
        const bool  posm = same && (v < 1.0f - F_EPS);
        if (!same && (v + F_MARGIN > minPos)) {
            negSum += expf(F_SNEG * (v - F_THRESH));
            any |= 2;
        }
        if (posm && (v - F_MARGIN < maxNeg)) {
            posSum += expf(-F_SPOS * (v - F_THRESH));
            any |= 1;
        }
    }
    redA[tid] = posSum;
    redB[tid] = negSum;
    redI[tid] = any;
    __syncthreads();
    for (int st = 128; st > 0; st >>= 1) {
        if (tid < st) {
            redA[tid] += redA[tid + st];
            redB[tid] += redB[tid + st];
            redI[tid] |= redI[tid + st];
        }
        __syncthreads();
    }
    if (tid == 0 && redI[0] == 3) {
        const float l = log1pf(redA[0]) / F_SPOS + log1pf(redB[0]) / F_SNEG;
        atomicAdd(out, l / (float)BATCH);
    }
}

// ---------------------------------------------------------------------------
// Launch
// ---------------------------------------------------------------------------
extern "C" void kernel_launch(void* const* d_in, const int* in_sizes, int n_in,
                              void* d_out, int out_size, void* d_ws, size_t ws_size,
                              hipStream_t stream) {
    (void)in_sizes; (void)n_in; (void)out_size; (void)ws_size;

    const float*     feats  = (const float*)d_in[0];
    const long long* labels = (const long long*)d_in[1];
    float*           out    = (float*)d_out;

    _Float16* Fh  = (_Float16*)d_ws;                               // 8 MB
    float*    sim = (float*)((char*)d_ws +
                             (size_t)BATCH * DIM * sizeof(_Float16)); // +64 MB

    ms_zero_out<<<1, 32, 0, stream>>>(out);
    ms_normalize_f16<<<BATCH, 256, 0, stream>>>(feats, Fh);
    ms_sim_gemm<<<dim3(BATCH / 64, 2), 256, 0, stream>>>(Fh, sim);
    ms_row_loss<<<BATCH, 256, 0, stream>>>(sim, labels, out);
}
